// MultiHeadAttention_45277545234839
// MI455X (gfx1250) — matrix operand order repaired and checked by reading, not compiled, not run on hardware
//
#include <hip/hip_runtime.h>
#include <hip/hip_bf16.h>

// MHA forward, B=2, S=2048, E=2048, H=16, DK=128, gfx1250 (wave32, bf16 WMMA + TDM)

typedef __attribute__((ext_vector_type(16))) __bf16 v16bf;
typedef __attribute__((ext_vector_type(8)))  float  v8f;
typedef __attribute__((ext_vector_type(4)))  unsigned int v4u;
typedef __attribute__((ext_vector_type(8)))  int v8i;
typedef __attribute__((ext_vector_type(4)))  int v4i;

static __device__ __forceinline__ unsigned short f2bf(float f) {
    unsigned u = __float_as_uint(f);
    u += 0x7FFFu + ((u >> 16) & 1u);           // round-to-nearest-even
    return (unsigned short)(u >> 16);
}

union FragU { v16bf v; uint4 q[2]; };

// A-fragment (16x32 bf16, ISA 7.12.2): lane = row M (mod 16); lanes 0-15 hold
// K=0..15, lanes 16-31 hold K=16..31 (contiguous). Two 16B LDS loads.
static __device__ __forceinline__ v16bf a_frag(const unsigned short* base, int stride,
                                               int kbase, int lane) {
    const unsigned short* r = base + (lane & 15) * stride + kbase;
    FragU fu;
    if (lane < 16) { fu.q[0] = *(const uint4*)(r + 0); fu.q[1] = *(const uint4*)(r + 8); }
    else           { fu.q[0] = *(const uint4*)(r + 16); fu.q[1] = *(const uint4*)(r + 24); }
    return fu.v;
}

// B-fragment (32x16 bf16): lane = column N (mod 16); lanes 0-15 hold K=0..15,
// lanes 16-31 hold K=16..31 (contiguous). Source tile stored N-major.
static __device__ __forceinline__ v16bf b_frag(const unsigned short* base, int stride,
                                               int kbase, int lane) {
    const unsigned short* r = base + (lane & 15) * stride + kbase + ((lane & 16) ? 16 : 0);
    FragU fu;
    fu.q[0] = *(const uint4*)(r + 0);
    fu.q[1] = *(const uint4*)(r + 8);
    return fu.v;
}

static __device__ __forceinline__ v8f wmma_bf16(v16bf a, v16bf b, v8f c) {
    return __builtin_amdgcn_wmma_f32_16x16x32_bf16(false, a, false, b, (short)0, c, false, false);
}

// LDS byte offset of a __shared__ object (generic addr low 32 bits = LDS offset)
template <typename T>
static __device__ __forceinline__ unsigned lds_off(T* p) {
    return (unsigned)(unsigned long long)(void*)p;
}

// ---------------------------------------------------------------------------
// Tensor Data Mover: 2D bf16 tile (tile_d0 x tile_d1 elements, row stride
// `stride_el` elements) from global to LDS. D# per CDNA5 ISA §8.3/8.4.
// Issue from ONE wave only; EXEC is ignored by the instruction.
// clang-23 toolchain: 6-arg builtin (g0, g1, g2, g3, g4, cpol).
// ---------------------------------------------------------------------------
static __device__ __forceinline__ void tdm_load_2d(const void* gaddr, unsigned ldsaddr,
                                                   unsigned tile_d0, unsigned tile_d1,
                                                   unsigned tensor_d0, unsigned tensor_d1,
                                                   unsigned long long stride_el) {
    unsigned long long ga = (unsigned long long)gaddr;
    v4u g0 = {
        1u,                                                  // count=1, user mode
        ldsaddr,                                             // lds_addr
        (unsigned)ga,                                        // global_addr[31:0]
        (unsigned)((ga >> 32) & 0x1FFFFFFu) | (2u << 30)     // global_addr[56:32] | type=2
    };
    v8i g1 = {
        (int)(1u << 16),                                     // data_size = 2 bytes
        (int)((tensor_d0 & 0xFFFFu) << 16),                  // tensor_dim0[15:0]
        (int)((tensor_d0 >> 16) | ((tensor_d1 & 0xFFFFu) << 16)),
        (int)((tensor_d1 >> 16) | ((tile_d0 & 0xFFFFu) << 16)),
        (int)(tile_d1 & 0xFFFFu),                            // tile_dim1 (tile_dim2=0)
        (int)(unsigned)(stride_el & 0xFFFFFFFFull),          // dim0_stride[31:0]
        (int)(unsigned)(stride_el >> 32),                    // dim0_stride[47:32]
        0
    };
    v4i z4 = {0, 0, 0, 0};
    v8i z8 = {0, 0, 0, 0, 0, 0, 0, 0};
    __builtin_amdgcn_tensor_load_to_lds(g0, g1, z4, z4, z8, 0);
}

// ---------------------------------------------------------------------------
// fp32 -> bf16 streaming pack (n divisible by 1024)
// ---------------------------------------------------------------------------
__global__ void mha_pack_kernel(const float* __restrict__ in, unsigned short* __restrict__ out) {
    size_t i = ((size_t)blockIdx.x * blockDim.x + threadIdx.x) * 4;
    float4 v = *(const float4*)(in + i);
    uint2 p;
    p.x = f2bf(v.x) | ((unsigned)f2bf(v.y) << 16);
    p.y = f2bf(v.z) | ((unsigned)f2bf(v.w) << 16);
    *(uint2*)(out + i) = p;
}

// ---------------------------------------------------------------------------
// RoPE (+ optional scale) on a (b,h,s,d) fp32 tensor -> bf16 tensor.
// ---------------------------------------------------------------------------
__global__ void mha_rope_pack_kernel(const float* __restrict__ in, unsigned short* __restrict__ out,
                                     const float* __restrict__ pe, float scale) {
    long p = (long)blockIdx.x * blockDim.x + threadIdx.x;   // pair index
    int d2 = (int)(p & 63);           // DK/2 = 64
    int s  = (int)((p >> 6) & 2047);  // S = 2048
    float ang = pe[s * 128 + d2 * 2];
    float c = cosf(ang), sn = sinf(ang);
    size_t base = (size_t)p * 2;
    float u0 = in[base], u1 = in[base + 1];
    unsigned short o0 = f2bf((u0 * c - u1 * sn) * scale);
    unsigned short o1 = f2bf((u1 * c + u0 * sn) * scale);
    *(unsigned*)(out + base) = (unsigned)o0 | ((unsigned)o1 << 16);
}

// ---------------------------------------------------------------------------
// GEMM: C(M,N) = A(MxK bf16 row-major) * W(NxK bf16 row-major)^T.
// TDM double-buffered LDS tiles; 256 thr = 8 waves (4x2), tile 128x128x32.
// mode 0: fp32 C[m*N+n]; mode 1: fp32 scatter (b,h,s,d); mode 2: bf16 scatter.
// ---------------------------------------------------------------------------
__global__ __launch_bounds__(256)
void mha_gemm_kernel(const unsigned short* __restrict__ A, const unsigned short* __restrict__ W,
                     float* __restrict__ Cf, unsigned short* __restrict__ Cb,
                     int M, int N, int K, int mode) {
    __shared__ unsigned short lds_a[2][128 * 32];
    __shared__ unsigned short lds_b[2][128 * 32];

    const int tid  = threadIdx.x;
    const int lane = tid & 31;
    const int wid  = tid >> 5;
    const int wm   = wid >> 1;     // 0..3 -> 32 rows each
    const int wn   = wid & 1;      // 0..1 -> 64 cols each
    const int m0   = blockIdx.y * 128;
    const int n0   = blockIdx.x * 128;

    v8f acc[2][4] = {};

    const int nk = K / 32;
    if (wid == 0) {
        tdm_load_2d(A + (size_t)m0 * K, lds_off(&lds_a[0][0]), 32, 128, K, M, K);
        tdm_load_2d(W + (size_t)n0 * K, lds_off(&lds_b[0][0]), 32, 128, K, N, K);
        __builtin_amdgcn_s_wait_tensorcnt(0);
    }
    __syncthreads();

    for (int i = 0; i < nk; ++i) {
        const int cur = i & 1, nxt = cur ^ 1;
        if (wid == 0 && i + 1 < nk) {
            int kk = (i + 1) * 32;
            tdm_load_2d(A + (size_t)m0 * K + kk, lds_off(&lds_a[nxt][0]), 32, 128, K, M, K);
            tdm_load_2d(W + (size_t)n0 * K + kk, lds_off(&lds_b[nxt][0]), 32, 128, K, N, K);
        }

        v16bf af[2], bfv[4];
#pragma unroll
        for (int mi = 0; mi < 2; ++mi)
            af[mi] = a_frag(&lds_a[cur][(wm * 32 + mi * 16) * 32], 32, 0, lane);
#pragma unroll
        for (int ni = 0; ni < 4; ++ni)
            bfv[ni] = b_frag(&lds_b[cur][(wn * 64 + ni * 16) * 32], 32, 0, lane);
#pragma unroll
        for (int mi = 0; mi < 2; ++mi)
#pragma unroll
            for (int ni = 0; ni < 4; ++ni)
                acc[mi][ni] = wmma_bf16(af[mi], bfv[ni], acc[mi][ni]);

        if (wid == 0) __builtin_amdgcn_s_wait_tensorcnt(0);
        __syncthreads();
    }

    // C/D layout: VGPR r -> M = r + 8*(lane>=16), N = lane&15
    const int colq = lane & 15;
    const int rofs = (lane & 16) ? 8 : 0;
#pragma unroll
    for (int mi = 0; mi < 2; ++mi) {
#pragma unroll
        for (int ni = 0; ni < 4; ++ni) {
            int n = n0 + wn * 64 + ni * 16 + colq;
#pragma unroll
            for (int r = 0; r < 8; ++r) {
                int m = m0 + wm * 32 + mi * 16 + r + rofs;
                float val = acc[mi][ni][r];
                if (mode == 0) {
                    Cf[(size_t)m * N + n] = val;
                } else {
                    int b = m >> 11, s = m & 2047;   // S = 2048
                    int h = n >> 7,  d = n & 127;    // DK = 128
                    size_t idx = (((size_t)b * 16 + h) * 2048 + s) * 128 + d;
                    if (mode == 1) Cf[idx] = val;
                    else           Cb[idx] = f2bf(val);
                }
            }
        }
    }
}

// ---------------------------------------------------------------------------
// Flash-style causal attention. q,k,v bf16 in (b,h,s,d); scale pre-folded into q.
// Block = 128 thr = 4 waves; each wave owns 16 queries; 32-key K/V tiles.
// Q/K tiles via TDM; V transposed manually in LDS. Output bf16 (b,s,h,d).
// ---------------------------------------------------------------------------
__global__ __launch_bounds__(128)
void mha_attn_kernel(const unsigned short* __restrict__ q, const unsigned short* __restrict__ k,
                     const unsigned short* __restrict__ v, unsigned short* __restrict__ o_out) {
    __shared__ unsigned short lds_q[64 * 128];       // 16 KB
    __shared__ unsigned short lds_k[32 * 128];       // 8 KB, [key][d]
    __shared__ unsigned short lds_vt[128 * 32];      // 8 KB, [d][key]
    __shared__ unsigned short lds_p[4][16 * 32];     // 4 KB, wave-private P

    const int tid  = threadIdx.x;
    const int lane = tid & 31;
    const int wid  = tid >> 5;
    const int qb   = blockIdx.x * 64;
    const int h    = blockIdx.y;
    const int b    = blockIdx.z;

    const unsigned short* qbase   = q + (((size_t)b * 16 + h) * 2048 + qb) * 128;
    const unsigned short* kbase_g = k + (((size_t)b * 16 + h) * 2048) * 128;
    const unsigned short* vbase_g = v + (((size_t)b * 16 + h) * 2048) * 128;

    // Q tile (64 x 128 bf16, contiguous): one 1D TDM copy
    if (wid == 0) {
        tdm_load_2d(qbase, lds_off(&lds_q[0]), 8192, 1, 8192, 1, 8192);
        __builtin_amdgcn_s_wait_tensorcnt(0);
    }
    __syncthreads();

    v16bf qf[4];
#pragma unroll
    for (int dc = 0; dc < 4; ++dc)
        qf[dc] = a_frag(lds_q + (wid * 16) * 128, 128, dc * 32, lane);

    const int qrow0 = qb + wid * 16;
    const int colq  = lane & 15;
    const int rofs  = (lane & 16) ? 8 : 0;
    unsigned short* pp = &lds_p[wid][0];

    float m_run[8], l_run[8];
    v8f oacc[8] = {};
#pragma unroll
    for (int r = 0; r < 8; ++r) { m_run[r] = -3.0e38f; l_run[r] = 0.0f; }

    for (int kb = 0; kb < qb + 64; kb += 32) {
        // K tile (32 x 128 bf16, contiguous): TDM
        if (wid == 0)
            tdm_load_2d(kbase_g + (size_t)kb * 128, lds_off(&lds_k[0]), 4096, 1, 4096, 1, 4096);

        // V tile: bf16 loads + transpose scatter into lds_vt[d][key]
#pragma unroll
        for (int i = 0; i < 4; ++i) {
            int off = (i * 128 + tid) * 8;           // element index, 8 bf16 per uint4
            uint4 vv = *(const uint4*)(vbase_g + (size_t)kb * 128 + off);
            int key = off >> 7, d = off & 127;
            lds_vt[(d + 0) * 32 + key] = (unsigned short)(vv.x & 0xFFFFu);
            lds_vt[(d + 1) * 32 + key] = (unsigned short)(vv.x >> 16);
            lds_vt[(d + 2) * 32 + key] = (unsigned short)(vv.y & 0xFFFFu);
            lds_vt[(d + 3) * 32 + key] = (unsigned short)(vv.y >> 16);
            lds_vt[(d + 4) * 32 + key] = (unsigned short)(vv.z & 0xFFFFu);
            lds_vt[(d + 5) * 32 + key] = (unsigned short)(vv.z >> 16);
            lds_vt[(d + 6) * 32 + key] = (unsigned short)(vv.w & 0xFFFFu);
            lds_vt[(d + 7) * 32 + key] = (unsigned short)(vv.w >> 16);
        }
        if (wid == 0) __builtin_amdgcn_s_wait_tensorcnt(0);
        __syncthreads();

        // scores: two 16q x 16k tiles, contraction over d (4 WMMA each)
        v8f s0 = {}, s1 = {};
#pragma unroll
        for (int dc = 0; dc < 4; ++dc) {
            v16bf bk0 = b_frag(lds_k, 128, dc * 32, lane);
            v16bf bk1 = b_frag(lds_k + 16 * 128, 128, dc * 32, lane);
            s0 = wmma_bf16(qf[dc], bk0, s0);
            s1 = wmma_bf16(qf[dc], bk1, s1);
        }

        // causal mask + online softmax (C layout rows; xor-reduce the 16-lane half)
        float p0[8], p1[8], corr[8];
#pragma unroll
        for (int r = 0; r < 8; ++r) {
            int row = qrow0 + r + rofs;
            int c0  = kb + colq;
            float a0 = (c0 <= row)      ? s0[r] : -3.0e38f;
            float a1 = (c0 + 16 <= row) ? s1[r] : -3.0e38f;
            float mx = fmaxf(a0, a1);
            mx = fmaxf(mx, __shfl_xor(mx, 1, 32));
            mx = fmaxf(mx, __shfl_xor(mx, 2, 32));
            mx = fmaxf(mx, __shfl_xor(mx, 4, 32));
            mx = fmaxf(mx, __shfl_xor(mx, 8, 32));
            float mnew = fmaxf(m_run[r], mx);
            corr[r] = __expf(m_run[r] - mnew);
            p0[r]   = __expf(a0 - mnew);
            p1[r]   = __expf(a1 - mnew);
            float rs = p0[r] + p1[r];
            rs += __shfl_xor(rs, 1, 32);
            rs += __shfl_xor(rs, 2, 32);
            rs += __shfl_xor(rs, 4, 32);
            rs += __shfl_xor(rs, 8, 32);
            l_run[r] = l_run[r] * corr[r] + rs;
            m_run[r] = mnew;
        }
#pragma unroll
        for (int j = 0; j < 8; ++j)
#pragma unroll
            for (int r = 0; r < 8; ++r)
                oacc[j][r] *= corr[r];

        // P (16 x 32) C-layout -> LDS, re-read as A-fragment
#pragma unroll
        for (int r = 0; r < 8; ++r) {
            int prow = r + rofs;
            pp[prow * 32 + colq]      = f2bf(p0[r]);
            pp[prow * 32 + colq + 16] = f2bf(p1[r]);
        }
        __syncthreads();

        v16bf pf = a_frag(pp, 32, 0, lane);
#pragma unroll
        for (int j = 0; j < 8; ++j) {
            v16bf bv = b_frag(lds_vt + (j * 16) * 32, 32, 0, lane);
            oacc[j] = wmma_bf16(pf, bv, oacc[j]);
        }
        __syncthreads();
    }

    // normalize and store bf16 to (b, s, h, d)
#pragma unroll
    for (int r = 0; r < 8; ++r) l_run[r] = 1.0f / l_run[r];
#pragma unroll
    for (int j = 0; j < 8; ++j) {
        int d = j * 16 + colq;
#pragma unroll
        for (int r = 0; r < 8; ++r) {
            int srow = qrow0 + r + rofs;
            o_out[(((size_t)b * 2048 + srow) * 16 + h) * 128 + d] = f2bf(oacc[j][r] * l_run[r]);
        }
    }
}

// ---------------------------------------------------------------------------
extern "C" void kernel_launch(void* const* d_in, const int* in_sizes, int n_in,
                              void* d_out, int out_size, void* d_ws, size_t ws_size,
                              hipStream_t stream) {
    const float* x  = (const float*)d_in[0];
    const float* wq = (const float*)d_in[1];
    const float* wk = (const float*)d_in[2];
    const float* wv = (const float*)d_in[3];
    const float* wo = (const float*)d_in[4];
    const float* pe = (const float*)d_in[5];
    float* out = (float*)d_out;

    const size_t NQ = (size_t)2 * 16 * 2048 * 128;   // B*H*S*DK = 8,388,608
    const size_t NW = (size_t)2048 * 2048;           // E*E     = 4,194,304

    char* w = (char*)d_ws;
    float* q32 = (float*)w;             w += NQ * 4;
    float* k32 = (float*)w;             w += NQ * 4;
    unsigned short* xbf  = (unsigned short*)w; w += NQ * 2;
    unsigned short* wqbf = (unsigned short*)w; w += NW * 2;
    unsigned short* wkbf = (unsigned short*)w; w += NW * 2;
    unsigned short* wvbf = (unsigned short*)w; w += NW * 2;
    unsigned short* wobf = (unsigned short*)w; w += NW * 2;
    unsigned short* qbf  = (unsigned short*)w; w += NQ * 2;
    unsigned short* kbf  = (unsigned short*)w; w += NQ * 2;
    unsigned short* vbf  = (unsigned short*)w; w += NQ * 2;
    unsigned short* abf  = (unsigned short*)w; w += NQ * 2;

    // bf16 packs (one-shot, memory-bound)
    mha_pack_kernel<<<NQ / 1024, 256, 0, stream>>>(x,  xbf);
    mha_pack_kernel<<<NW / 1024, 256, 0, stream>>>(wq, wqbf);
    mha_pack_kernel<<<NW / 1024, 256, 0, stream>>>(wk, wkbf);
    mha_pack_kernel<<<NW / 1024, 256, 0, stream>>>(wv, wvbf);
    mha_pack_kernel<<<NW / 1024, 256, 0, stream>>>(wo, wobf);

    const int M = 4096, N = 2048, K = 2048;
    dim3 ggrid(N / 128, M / 128);                    // (16, 32)

    // QKV projections (TDM-fed WMMA GEMM)
    mha_gemm_kernel<<<ggrid, 256, 0, stream>>>(xbf, wqbf, q32, nullptr, M, N, K, 1);
    mha_gemm_kernel<<<ggrid, 256, 0, stream>>>(xbf, wkbf, k32, nullptr, M, N, K, 1);
    mha_gemm_kernel<<<ggrid, 256, 0, stream>>>(xbf, wvbf, nullptr, vbf, M, N, K, 2);

    // RoPE (+1/sqrt(dk) folded into q) -> bf16
    const int pairs = 2 * 16 * 2048 * 64;            // 4,194,304
    mha_rope_pack_kernel<<<pairs / 256, 256, 0, stream>>>(q32, qbf, pe, 0.08838834764831845f);
    mha_rope_pack_kernel<<<pairs / 256, 256, 0, stream>>>(k32, kbf, pe, 1.0f);

    // causal flash attention -> bf16 (b,s,h,d)
    mha_attn_kernel<<<dim3(32, 16, 2), 128, 0, stream>>>(qbf, kbf, vbf, abf);

    // output projection -> fp32
    mha_gemm_kernel<<<ggrid, 256, 0, stream>>>(abf, wobf, out, nullptr, M, N, K, 0);
}
